// MoELayer_91104846283134
// MI455X (gfx1250) — matrix operand order repaired and checked by reading, not compiled, not run on hardware
//
#include <hip/hip_runtime.h>
#include <hip/hip_bf16.h>

typedef unsigned short u16;
typedef unsigned int u32;

#define DIM 1024
#define NEXP 8

typedef __bf16 v16bf __attribute__((ext_vector_type(16)));
typedef float  v8f   __attribute__((ext_vector_type(8)));

union Frag32B { uint4 u[2]; v16bf v; };

__device__ __forceinline__ u16 f2bf(float f) {
    u32 u = __float_as_uint(f);
    u += 0x7fffu + ((u >> 16) & 1u);       // round-to-nearest-even
    return (u16)(u >> 16);
}

__device__ __forceinline__ void load_frag(Frag32B& f, const u16* p, int stride_hi) {
    // stride_hi = 16 for B-frags / 16-elem split; A uses two chunks at +0 and +16
    f.u[0] = *(const uint4*)(p);
    f.u[1] = *(const uint4*)(p + stride_hi);
}

// ---------------- x -> bf16 ----------------
__global__ void k_convert_x(const float* __restrict__ x, u16* __restrict__ xb, int n) {
    int i = blockIdx.x * 256 + threadIdx.x;
    if (i < n) xb[i] = f2bf(x[i]);
}

// ---- fp32 [E][K][N] -> bf16 [E][N][K] (transposed) for all 3 weight sets ----
__global__ void k_transpose_w(const float* __restrict__ wg, const float* __restrict__ wu,
                              const float* __restrict__ wd,
                              u16* __restrict__ tg, u16* __restrict__ tu, u16* __restrict__ td) {
    __shared__ float tile[32][33];
    int z = blockIdx.z;                       // 0..23
    const float* src = (z < 8) ? wg : (z < 16) ? wu : wd;
    u16* dst = (z < 8) ? tg : (z < 16) ? tu : td;
    int e = z & 7;
    int k0 = blockIdx.y * 32, n0 = blockIdx.x * 32;
    int tx = threadIdx.x & 31, ty = threadIdx.x >> 5;   // 32 x 8
    const float* s = src + (size_t)e * DIM * DIM;
    u16* d = dst + (size_t)e * DIM * DIM;
#pragma unroll
    for (int i = 0; i < 32; i += 8)
        tile[ty + i][tx] = s[(size_t)(k0 + ty + i) * DIM + n0 + tx];
    __syncthreads();
#pragma unroll
    for (int i = 0; i < 32; i += 8)
        d[(size_t)(n0 + ty + i) * DIM + k0 + tx] = f2bf(tile[tx][ty + i]);
}

// ---------------- router: one wave per token ----------------
__global__ void k_router(const float* __restrict__ x, const float* __restrict__ rw,
                         const float* __restrict__ rb, int* __restrict__ counts,
                         int* __restrict__ btok, float* __restrict__ bwgt, int T) {
    int lane = threadIdx.x & 31;
    int wv   = threadIdx.x >> 5;
    int t = blockIdx.x * 8 + wv;
    if (t >= T) return;

    float acc[NEXP];
#pragma unroll
    for (int e = 0; e < NEXP; ++e) acc[e] = 0.0f;

    const float* xr = x + (size_t)t * DIM;
    for (int d = lane; d < DIM; d += 32) {
        float xv = xr[d];
        const float* r = rw + (size_t)d * NEXP;
#pragma unroll
        for (int e = 0; e < NEXP; ++e) acc[e] += xv * r[e];
    }
#pragma unroll
    for (int e = 0; e < NEXP; ++e) {
#pragma unroll
        for (int off = 16; off > 0; off >>= 1)
            acc[e] += __shfl_xor(acc[e], off, 32);
    }
    if (lane == 0) {
#pragma unroll
        for (int e = 0; e < NEXP; ++e) acc[e] += rb[e];
        float v1 = -3.4e38f; int i1 = 0;
#pragma unroll
        for (int e = 0; e < NEXP; ++e) if (acc[e] > v1) { v1 = acc[e]; i1 = e; }
        float v2 = -3.4e38f; int i2 = (i1 == 0) ? 1 : 0;
#pragma unroll
        for (int e = 0; e < NEXP; ++e) if (e != i1 && acc[e] > v2) { v2 = acc[e]; i2 = e; }
        float e2  = __expf(v2 - v1);
        float inv = 1.0f / (1.0f + e2);
        int s1 = atomicAdd(&counts[i1], 1);
        btok[i1 * T + s1] = t;  bwgt[i1 * T + s1] = inv;
        int s2 = atomicAdd(&counts[i2], 1);
        btok[i2 * T + s2] = t;  bwgt[i2 * T + s2] = e2 * inv;
    }
}

__global__ void k_prefix(const int* __restrict__ counts, int* __restrict__ offs) {
    if (threadIdx.x == 0) {
        int a = 0;
#pragma unroll
        for (int e = 0; e < NEXP; ++e) { offs[e] = a; a += counts[e]; }
    }
}

// -------- GEMM1: gathered tokens x (gate, up), fused SiLU, bf16 out --------
// WG = 8 waves, layout 2(M) x 4(N); each wave owns a 32x32 C tile (2x2 frags)
// for BOTH gate and up -> 8 WMMAs per 12 b128 loads per k-step.
__global__ __launch_bounds__(256)
void k_gemm_gateup(const u16* __restrict__ xb, const u16* __restrict__ wtg,
                   const u16* __restrict__ wtu, const int* __restrict__ counts,
                   const int* __restrict__ offs, const int* __restrict__ btok,
                   u16* __restrict__ act, int T) {
    const int e = blockIdx.z;
    const int count = counts[e];
    const int tile_s = blockIdx.y * 64;
    if (tile_s >= count) return;

    const int lane = threadIdx.x & 31;
    const int wv   = threadIdx.x >> 5;     // 8 waves
    const int mi   = wv & 1;               // 2 M sub-tiles of 32
    const int ni   = wv >> 1;              // 4 N sub-tiles of 32
    const int l15  = lane & 15;
    const int hi   = lane >> 4;

    // two A row-fragments (M and M+16)
    const u16* arow[2];
    int srow[2];
#pragma unroll
    for (int ma = 0; ma < 2; ++ma) {
        int s = tile_s + mi * 32 + ma * 16 + l15;
        srow[ma] = s;
        int sidx = (s < count) ? s : (count - 1);
        int tok = btok[e * T + sidx];
        arow[ma] = xb + (size_t)tok * DIM + hi * 8;    // A: hi lanes at k+8
    }

    // two B col-fragments per weight matrix (N and N+16)
    const int nbase = blockIdx.x * 128 + ni * 32;
    const u16* bg[2]; const u16* bu[2];
#pragma unroll
    for (int nb = 0; nb < 2; ++nb) {
        int ncol = nbase + nb * 16 + l15;
        bg[nb] = wtg + ((size_t)e * DIM + ncol) * DIM + hi * 8;   // B: hi lanes at k+8
        bu[nb] = wtu + ((size_t)e * DIM + ncol) * DIM + hi * 8;
    }

    v8f accg[2][2] = {}; v8f accu[2][2] = {};
    for (int kt = 0; kt < DIM; kt += 32) {
        __builtin_prefetch(bg[0] + kt + 256, 0, 1);
        __builtin_prefetch(bu[0] + kt + 256, 0, 1);
        Frag32B A[2], Bg[2], Bu[2];
#pragma unroll
        for (int ma = 0; ma < 2; ++ma) load_frag(A[ma], arow[ma] + kt, 16);
#pragma unroll
        for (int nb = 0; nb < 2; ++nb) {
            load_frag(Bg[nb], bg[nb] + kt, 16);
            load_frag(Bu[nb], bu[nb] + kt, 16);
        }
#pragma unroll
        for (int ma = 0; ma < 2; ++ma)
#pragma unroll
            for (int nb = 0; nb < 2; ++nb) {
                accg[ma][nb] = __builtin_amdgcn_wmma_f32_16x16x32_bf16(
                    false, A[ma].v, false, Bg[nb].v, (short)0, accg[ma][nb], false, false);
                accu[ma][nb] = __builtin_amdgcn_wmma_f32_16x16x32_bf16(
                    false, A[ma].v, false, Bu[nb].v, (short)0, accu[ma][nb], false, false);
            }
    }

    const int off_e = offs[e];
#pragma unroll
    for (int ma = 0; ma < 2; ++ma)
#pragma unroll
        for (int nb = 0; nb < 2; ++nb) {
            int ncol = nbase + nb * 16 + l15;
#pragma unroll
            for (int r = 0; r < 8; ++r) {
                int m = r + hi * 8;
                int s = tile_s + mi * 32 + ma * 16 + m;
                if (s < count) {
                    float g = accg[ma][nb][r], u = accu[ma][nb][r];
                    float a = (g / (1.0f + __expf(-g))) * u;        // SiLU(g) * u
                    act[(size_t)(off_e + s) * DIM + ncol] = f2bf(a);
                }
            }
        }
}

// -------- GEMM2: activated x down, scale by router weight, atomic combine --------
__global__ __launch_bounds__(256)
void k_gemm_down(const u16* __restrict__ act, const u16* __restrict__ wtd,
                 const int* __restrict__ counts, const int* __restrict__ offs,
                 const int* __restrict__ btok, const float* __restrict__ bwgt,
                 float* __restrict__ out, int T) {
    const int e = blockIdx.z;
    const int count = counts[e];
    const int tile_s = blockIdx.y * 64;
    if (tile_s >= count) return;

    const int lane = threadIdx.x & 31;
    const int wv   = threadIdx.x >> 5;
    const int mi   = wv & 1;
    const int ni   = wv >> 1;
    const int l15  = lane & 15;
    const int hi   = lane >> 4;
    const int off_e = offs[e];

    const u16* arow[2];
    int mytok[2]; float myw[2];
#pragma unroll
    for (int ma = 0; ma < 2; ++ma) {
        int s = tile_s + mi * 32 + ma * 16 + l15;
        int sidx = (s < count) ? s : (count - 1);
        mytok[ma] = btok[e * T + sidx];
        myw[ma]   = bwgt[e * T + sidx];
        arow[ma]  = act + (size_t)(off_e + s) * DIM + hi * 8;
    }

    const int nbase = blockIdx.x * 128 + ni * 32;
    const u16* bd[2];
#pragma unroll
    for (int nb = 0; nb < 2; ++nb) {
        int ncol = nbase + nb * 16 + l15;
        bd[nb] = wtd + ((size_t)e * DIM + ncol) * DIM + hi * 8;
    }

    v8f acc[2][2] = {};
    for (int kt = 0; kt < DIM; kt += 32) {
        __builtin_prefetch(bd[0] + kt + 256, 0, 1);
        Frag32B A[2], B[2];
#pragma unroll
        for (int ma = 0; ma < 2; ++ma) load_frag(A[ma], arow[ma] + kt, 16);
#pragma unroll
        for (int nb = 0; nb < 2; ++nb) load_frag(B[nb], bd[nb] + kt, 16);
#pragma unroll
        for (int ma = 0; ma < 2; ++ma)
#pragma unroll
            for (int nb = 0; nb < 2; ++nb)
                acc[ma][nb] = __builtin_amdgcn_wmma_f32_16x16x32_bf16(
                    false, A[ma].v, false, B[nb].v, (short)0, acc[ma][nb], false, false);
    }

#pragma unroll
    for (int ma = 0; ma < 2; ++ma)
#pragma unroll
        for (int nb = 0; nb < 2; ++nb) {
            int ncol = nbase + nb * 16 + l15;
#pragma unroll
            for (int r = 0; r < 8; ++r) {
                int m = r + hi * 8;
                int s = tile_s + mi * 32 + ma * 16 + m;
                float w  = __shfl(myw[ma], m, 32);
                int   tk = __shfl(mytok[ma], m, 32);
                if (s < count)
                    atomicAdd(&out[(size_t)tk * DIM + ncol], acc[ma][nb][r] * w);
            }
        }
}

extern "C" void kernel_launch(void* const* d_in, const int* in_sizes, int n_in,
                              void* d_out, int out_size, void* d_ws, size_t ws_size,
                              hipStream_t stream) {
    const float* x  = (const float*)d_in[0];
    const float* rw = (const float*)d_in[1];
    const float* rb = (const float*)d_in[2];
    const float* wg = (const float*)d_in[3];
    const float* wu = (const float*)d_in[4];
    const float* wd = (const float*)d_in[5];
    float* out = (float*)d_out;

    const int T = in_sizes[0] / DIM;      // 16384 tokens
    (void)n_in; (void)ws_size;

    char* ws = (char*)d_ws;
    size_t off = 0;
    auto carve = [&](size_t bytes) -> void* {
        void* p = ws + off;
        off = (off + bytes + 255) & ~(size_t)255;
        return p;
    };
    u16*   xb    = (u16*)  carve((size_t)T * DIM * 2);
    u16*   wtg   = (u16*)  carve((size_t)NEXP * DIM * DIM * 2);
    u16*   wtu   = (u16*)  carve((size_t)NEXP * DIM * DIM * 2);
    u16*   wtd   = (u16*)  carve((size_t)NEXP * DIM * DIM * 2);
    u16*   act   = (u16*)  carve(((size_t)2 * T + 128) * DIM * 2);
    int*   btok  = (int*)  carve((size_t)NEXP * T * 4);
    float* bwgt  = (float*)carve((size_t)NEXP * T * 4);
    int*   cnts  = (int*)  carve(256);
    int*   offs  = (int*)  carve(256);

    hipMemsetAsync(out, 0, (size_t)out_size * sizeof(float), stream);
    hipMemsetAsync(cnts, 0, NEXP * sizeof(int), stream);

    k_convert_x<<<(T * DIM) / 256, 256, 0, stream>>>(x, xb, T * DIM);
    k_transpose_w<<<dim3(DIM / 32, DIM / 32, 24), 256, 0, stream>>>(wg, wu, wd, wtg, wtu, wtd);
    k_router<<<(T + 7) / 8, 256, 0, stream>>>(x, rw, rb, cnts, btok, bwgt, T);
    k_prefix<<<1, 32, 0, stream>>>(cnts, offs);

    dim3 ggrid(DIM / 128, (T + 63) / 64, NEXP);
    k_gemm_gateup<<<ggrid, 256, 0, stream>>>(xb, wtg, wtu, cnts, offs, btok, act, T);
    k_gemm_down<<<ggrid, 256, 0, stream>>>(act, wtd, cnts, offs, btok, bwgt, out, T);
}
